// NPTLayer_3994319585427
// MI455X (gfx1250) — matrix operand order repaired and checked by reading, not compiled, not run on hardware
//
#include <hip/hip_runtime.h>
#include <math.h>

#define D_MODEL 2048
#define D_FFN   8192
#define RANK    16
#define NTOK    4096          // B*S = 2*2048
#define MOD_SCALE 0.1f
#define EPS 1e-6f

typedef __attribute__((ext_vector_type(16))) __bf16 v16bf;
typedef __attribute__((ext_vector_type(8)))  __bf16 v8bf;
typedef __attribute__((ext_vector_type(8)))  float  v8f;
typedef __attribute__((ext_vector_type(4)))  int    v4i;

// address-space qualified v4i for the async global->LDS builtin
typedef __attribute__((address_space(1))) v4i v4i_g;   // global (AS1)
typedef __attribute__((address_space(3))) v4i v4i_l;   // LDS    (AS3)

// ---- gfx1250 async global->LDS path (ASYNCcnt) -----------------------------
#if defined(__has_builtin)
#  if __has_builtin(__builtin_amdgcn_global_load_async_to_lds_b128) && \
      __has_builtin(__builtin_amdgcn_s_wait_asynccnt)
#    define USE_ASYNC 1
#  endif
#endif
#ifndef USE_ASYNC
#  define USE_ASYNC 0
#endif

#if USE_ASYNC
__device__ inline void async_b128(const __bf16* g, __bf16* l) {
    __builtin_amdgcn_global_load_async_to_lds_b128(
        (v4i_g*)(void*)g, (v4i_l*)(void*)l, 0, 0);
}
#endif

// ---------------------------------------------------------------------------
// Kernel 1: fp32 -> bf16 weight conversion (8 elements / thread, b128 loads)
// ---------------------------------------------------------------------------
__global__ __launch_bounds__(256) void cvt_f32_bf16(const float* __restrict__ src,
                                                    __bf16* __restrict__ dst, int n) {
    int idx = (blockIdx.x * 256 + threadIdx.x) * 8;
    if (idx >= n) return;
    float4 a = ((const float4*)(src + idx))[0];
    float4 b = ((const float4*)(src + idx))[1];
    float f[8] = {a.x, a.y, a.z, a.w, b.x, b.y, b.z, b.w};
    v8bf o;
#pragma unroll
    for (int j = 0; j < 8; ++j) o[j] = (__bf16)f[j];
    *(v8bf*)(dst + idx) = o;
}

// ---------------------------------------------------------------------------
// Kernel 2: fused RMSNorm + LoRA rank projections (one block per token)
// ---------------------------------------------------------------------------
__global__ __launch_bounds__(256) void rmsnorm_lora(const float* __restrict__ hidden,
                                                    const float* __restrict__ attn,
                                                    const float* __restrict__ A,
                                                    const float* __restrict__ gamma,
                                                    __bf16* __restrict__ hbf,
                                                    float* __restrict__ cOut) {
    __shared__ float red[8][34];
    __shared__ float fin[34];

    const int t    = blockIdx.x;
    const int tid  = threadIdx.x;
    const int wave = tid >> 5;
    const int lane = tid & 31;
    const int d0   = tid * 8;
    const long base = (long)t * D_MODEL;

    float4 xa = ((const float4*)(hidden + base + d0))[0];
    float4 xb = ((const float4*)(hidden + base + d0))[1];
    float4 ta = ((const float4*)(attn   + base + d0))[0];
    float4 tb = ((const float4*)(attn   + base + d0))[1];
    float4 ga = ((const float4*)(gamma  + d0))[0];
    float4 gb = ((const float4*)(gamma  + d0))[1];

    float x[8]  = {xa.x, xa.y, xa.z, xa.w, xb.x, xb.y, xb.z, xb.w};
    float at[8] = {ta.x, ta.y, ta.z, ta.w, tb.x, tb.y, tb.z, tb.w};
    float g[8]  = {ga.x, ga.y, ga.z, ga.w, gb.x, gb.y, gb.z, gb.w};

    float vals[33];
#pragma unroll
    for (int i = 0; i < 33; ++i) vals[i] = 0.f;

#pragma unroll
    for (int j = 0; j < 8; ++j) {
        int d = d0 + j;
        vals[0] += x[j] * x[j];
        float xg = x[j] * g[j];
        const float4* Ar = (const float4*)(A + d * RANK);
        float4 a0 = Ar[0], a1 = Ar[1], a2 = Ar[2], a3 = Ar[3];
        float av[16] = {a0.x,a0.y,a0.z,a0.w, a1.x,a1.y,a1.z,a1.w,
                        a2.x,a2.y,a2.z,a2.w, a3.x,a3.y,a3.z,a3.w};
#pragma unroll
        for (int r = 0; r < 16; ++r) {
            vals[1 + r]  += xg    * av[r];
            vals[17 + r] += at[j] * av[r];
        }
    }

#pragma unroll
    for (int i = 0; i < 33; ++i) {
        float v = vals[i];
        for (int off = 16; off > 0; off >>= 1) v += __shfl_down(v, off, 32);
        if (lane == 0) red[wave][i] = v;
    }
    __syncthreads();
    if (tid < 33) {
        float s = 0.f;
#pragma unroll
        for (int w2 = 0; w2 < 8; ++w2) s += red[w2][tid];
        fin[tid] = s;
    }
    __syncthreads();

    float rs = rsqrtf(fin[0] * (1.0f / D_MODEL) + EPS);

    v8bf hv;
#pragma unroll
    for (int j = 0; j < 8; ++j) hv[j] = (__bf16)(x[j] * g[j] * rs);
    *(v8bf*)(hbf + base + d0) = hv;

    if (tid < RANK) {
        float mod = MOD_SCALE * tanhf(fin[17 + tid]);
        cOut[t * RANK + tid] = rs * fin[1 + tid] * mod;
    }
}

// ---------------------------------------------------------------------------
// Fragment helpers (wave32 16-bit operands; contiguous k order used by this code)
//  A 16x32: lanes 0-15 -> row=l,   K 0..15 contiguous ; lanes 16-31 -> row=l-16, K 16..31
//  B 32x16: lanes 0-15 -> col=l,   K 0..15 contiguous ; lanes 16-31 -> col=l-16, K 16..31
// ---------------------------------------------------------------------------
__device__ inline v16bf load_a_frag(const __bf16* p) {
    v8bf lo = *(const v8bf*)(p);
    v8bf hi = *(const v8bf*)(p + 8);
    v16bf a;
#pragma unroll
    for (int i = 0; i < 8; ++i) { a[i] = lo[i]; a[i + 8] = hi[i]; }
    return a;
}

#define WMMA_BF16(a, b, c) \
    __builtin_amdgcn_wmma_f32_16x16x32_bf16(false, (a), false, (b), (short)0, (c), false, false)

// ---------------------------------------------------------------------------
// Kernel 3: fused gate/up GEMM + LoRA delta + SiLU.
//  8 waves; M_TILE=256 (32 rows/wave: 2 A-frags), N_TILE=64 (4 subtiles).
//  B tiles (Win+Wup, 8KB/K-step) staged in LDS via async loads, double
//  buffered.  A fragments double-buffered in registers across K iterations;
//  B fragments software-pipelined one subtile ahead.
// ---------------------------------------------------------------------------
__global__ __launch_bounds__(256) void ffn_gate_up(const __bf16* __restrict__ hbf,
                                                   const __bf16* __restrict__ win,
                                                   const __bf16* __restrict__ wup,
                                                   const float*  __restrict__ cIn,
                                                   const float*  __restrict__ Bmat,
                                                   __bf16* __restrict__ act) {
    __shared__ __attribute__((aligned(64))) __bf16 sB[2][2][64][32]; // [buf][mat][f][k]
    __shared__ float cT[256][RANK];
    __shared__ float bmT[RANK][64];

    const int tid = threadIdx.x;
    const int w   = tid >> 5;
    const int l   = tid & 31;
    const int hi  = l >> 4;
    const int n   = l & 15;
    const int blockM  = blockIdx.y * 256;
    const int colBase = blockIdx.x * 64;

    // epilogue tables
    for (int i = tid; i < 256 * RANK; i += 256) {
        int m = i >> 4, r = i & 15;
        cT[m][r] = cIn[(blockM + m) * RANK + r];
    }
    for (int i = tid; i < RANK * 64; i += 256) {
        int r = i >> 6, cc = i & 63;
        bmT[r][cc] = Bmat[r * D_FFN + colBase + cc];
    }

    // per-thread staging slot: 32B of one weight row segment
    const int sm    = tid >> 7;          // 0 = Win, 1 = Wup
    const int sidx  = tid & 127;
    const int srow  = sidx >> 1;         // 0..63 (local f)
    const int shalf = sidx & 1;
    const __bf16* gW = (sm ? wup : win) + (long)(colBase + srow) * D_MODEL + shalf * 16;
    __bf16* lW[2] = {&sB[0][sm][srow][shalf * 16], &sB[1][sm][srow][shalf * 16]};

    // pre-stage K-step 0
    {
        v8bf t0 = *(const v8bf*)(gW);
        v8bf t1 = *(const v8bf*)(gW + 8);
        *(v8bf*)(lW[0])     = t0;
        *(v8bf*)(lW[0] + 8) = t1;
    }

    const __bf16* aB0 = hbf + (long)(blockM + w * 32 + n)      * D_MODEL + hi * 16;
    const __bf16* aB1 = hbf + (long)(blockM + w * 32 + 16 + n) * D_MODEL + hi * 16;

    // A fragments double-buffered across K iterations
    v16bf a0 = load_a_frag(aB0);
    v16bf a1 = load_a_frag(aB1);

    v8f accg[2][4] = {};
    v8f accu[2][4] = {};

    int pb = 0;
    for (int kk = 0; kk < D_MODEL; kk += 32, pb ^= 1) {
#if USE_ASYNC
        __builtin_amdgcn_s_wait_asynccnt(0);
#endif
        __syncthreads();

        const bool hasNext = (kk + 32) < D_MODEL;
        v8bf s0, s1;
        if (hasNext) {
#if USE_ASYNC
            async_b128(gW + kk + 32,     lW[pb ^ 1]);
            async_b128(gW + kk + 32 + 8, lW[pb ^ 1] + 8);
#else
            s0 = *(const v8bf*)(gW + kk + 32);
            s1 = *(const v8bf*)(gW + kk + 32 + 8);
#endif
        }

        // issue next iteration's A loads early (consumed next iteration)
        v16bf a0n = a0, a1n = a1;
        if (hasNext) {
            a0n = load_a_frag(aB0 + kk + 32);
            a1n = load_a_frag(aB1 + kk + 32);
        }

        // B fragments pipelined one subtile ahead
        v16bf bg = *(const v16bf*)&sB[pb][0][n][hi * 16];
        v16bf bu = *(const v16bf*)&sB[pb][1][n][hi * 16];
#pragma unroll
        for (int s = 0; s < 4; ++s) {
            v16bf bgn = bg, bun = bu;
            if (s < 3) {
                bgn = *(const v16bf*)&sB[pb][0][(s + 1) * 16 + n][hi * 16];
                bun = *(const v16bf*)&sB[pb][1][(s + 1) * 16 + n][hi * 16];
            }
            accg[0][s] = WMMA_BF16(a0, bg, accg[0][s]);
            accg[1][s] = WMMA_BF16(a1, bg, accg[1][s]);
            accu[0][s] = WMMA_BF16(a0, bu, accu[0][s]);
            accu[1][s] = WMMA_BF16(a1, bu, accu[1][s]);
            bg = bgn; bu = bun;
        }

        a0 = a0n; a1 = a1n;

#if !USE_ASYNC
        if (hasNext) {
            *(v8bf*)(lW[pb ^ 1])     = s0;
            *(v8bf*)(lW[pb ^ 1] + 8) = s1;
        }
#endif
    }

    // epilogue: + rank-16 delta, SiLU, * up, bf16 store
#pragma unroll
    for (int f = 0; f < 2; ++f) {
#pragma unroll
        for (int s = 0; s < 4; ++s) {
            const int cl = s * 16 + n;
#pragma unroll
            for (int i = 0; i < 8; ++i) {
                const int mloc = w * 32 + f * 16 + i + 8 * hi;
                float delta = 0.f;
#pragma unroll
                for (int r = 0; r < RANK; ++r) delta += cT[mloc][r] * bmT[r][cl];
                float gv = accg[f][s][i] + delta;
                float sg = gv / (1.0f + __expf(-gv));
                float val = sg * accu[f][s][i];
                act[(long)(blockM + mloc) * D_FFN + colBase + cl] = (__bf16)val;
            }
        }
    }
}

// ---------------------------------------------------------------------------
// Kernel 4: down projection + residual.  out = residual + act @ Wdown^T
//  M=4096 (256/block), N=2048 (64/block), K=8192.  Same pipelining.
// ---------------------------------------------------------------------------
__global__ __launch_bounds__(256) void ffn_down(const __bf16* __restrict__ act,
                                                const __bf16* __restrict__ wd,
                                                const float*  __restrict__ residual,
                                                float* __restrict__ out) {
    __shared__ __attribute__((aligned(64))) __bf16 sB[2][64][32];   // [buf][d][k]

    const int tid = threadIdx.x;
    const int w   = tid >> 5;
    const int l   = tid & 31;
    const int hi  = l >> 4;
    const int n   = l & 15;
    const int blockM = blockIdx.y * 256;
    const int dBase  = blockIdx.x * 64;

    const int srow = tid >> 2;           // 0..63 (local d)
    const int sq   = tid & 3;            // 8-half chunk
    const __bf16* gW = wd + (long)(dBase + srow) * D_FFN + sq * 8;
    __bf16* lW[2] = {&sB[0][srow][sq * 8], &sB[1][srow][sq * 8]};

    {
        v8bf t0 = *(const v8bf*)(gW);
        *(v8bf*)(lW[0]) = t0;
    }

    const __bf16* aB0 = act + (long)(blockM + w * 32 + n)      * D_FFN + hi * 16;
    const __bf16* aB1 = act + (long)(blockM + w * 32 + 16 + n) * D_FFN + hi * 16;

    v16bf a0 = load_a_frag(aB0);
    v16bf a1 = load_a_frag(aB1);

    v8f acc[2][4] = {};

    int pb = 0;
    for (int kk = 0; kk < D_FFN; kk += 32, pb ^= 1) {
#if USE_ASYNC
        __builtin_amdgcn_s_wait_asynccnt(0);
#endif
        __syncthreads();

        const bool hasNext = (kk + 32) < D_FFN;
        v8bf s0;
        if (hasNext) {
#if USE_ASYNC
            async_b128(gW + kk + 32, lW[pb ^ 1]);
#else
            s0 = *(const v8bf*)(gW + kk + 32);
#endif
        }

        v16bf a0n = a0, a1n = a1;
        if (hasNext) {
            a0n = load_a_frag(aB0 + kk + 32);
            a1n = load_a_frag(aB1 + kk + 32);
        }

        v16bf b = *(const v16bf*)&sB[pb][n][hi * 16];
#pragma unroll
        for (int s = 0; s < 4; ++s) {
            v16bf bn = b;
            if (s < 3) bn = *(const v16bf*)&sB[pb][(s + 1) * 16 + n][hi * 16];
            acc[0][s] = WMMA_BF16(a0, b, acc[0][s]);
            acc[1][s] = WMMA_BF16(a1, b, acc[1][s]);
            b = bn;
        }

        a0 = a0n; a1 = a1n;

#if !USE_ASYNC
        if (hasNext) *(v8bf*)(lW[pb ^ 1]) = s0;
#endif
    }

#pragma unroll
    for (int f = 0; f < 2; ++f) {
#pragma unroll
        for (int s = 0; s < 4; ++s) {
            const int cl = s * 16 + n;
#pragma unroll
            for (int i = 0; i < 8; ++i) {
                const int mloc = w * 32 + f * 16 + i + 8 * hi;
                const long idx = (long)(blockM + mloc) * D_MODEL + dBase + cl;
                out[idx] = residual[idx] + acc[f][s][i];
            }
        }
    }
}

// ---------------------------------------------------------------------------
// Launcher
// ---------------------------------------------------------------------------
extern "C" void kernel_launch(void* const* d_in, const int* in_sizes, int n_in,
                              void* d_out, int out_size, void* d_ws, size_t ws_size,
                              hipStream_t stream) {
    const float* hidden = (const float*)d_in[0];
    const float* attn   = (const float*)d_in[1];
    const float* A      = (const float*)d_in[2];
    const float* Bmat   = (const float*)d_in[3];
    const float* Win    = (const float*)d_in[4];
    const float* Wup    = (const float*)d_in[5];
    const float* Wdn    = (const float*)d_in[6];
    const float* gamma  = (const float*)d_in[7];

    const long WNE = (long)D_FFN * D_MODEL;
    char* ws = (char*)d_ws;
    __bf16* winb = (__bf16*)(ws);
    __bf16* wupb = (__bf16*)(ws + WNE * 2);
    __bf16* wdnb = (__bf16*)(ws + WNE * 4);
    __bf16* hbf  = (__bf16*)(ws + WNE * 6);
    __bf16* actb = (__bf16*)(ws + WNE * 6 + (long)NTOK * D_MODEL * 2);
    float*  cbuf = (float*) (ws + WNE * 6 + (long)NTOK * D_MODEL * 2
                                            + (long)NTOK * D_FFN * 2);

    const int cvtBlocks = (int)(WNE / (256 * 8));
    cvt_f32_bf16<<<cvtBlocks, 256, 0, stream>>>(Win, winb, (int)WNE);
    cvt_f32_bf16<<<cvtBlocks, 256, 0, stream>>>(Wup, wupb, (int)WNE);
    cvt_f32_bf16<<<cvtBlocks, 256, 0, stream>>>(Wdn, wdnb, (int)WNE);

    rmsnorm_lora<<<NTOK, 256, 0, stream>>>(hidden, attn, A, gamma, hbf, cbuf);

    ffn_gate_up<<<dim3(D_FFN / 64, NTOK / 256), 256, 0, stream>>>(hbf, winb, wupb,
                                                                  cbuf, Bmat, actb);

    ffn_down<<<dim3(D_MODEL / 64, NTOK / 256), 256, 0, stream>>>(actb, wdnb, hidden,
                                                                 (float*)d_out);
}
